// MultiheadAttentionExpertModule_5231270166971
// MI455X (gfx1250) — matrix operand order repaired and checked by reading, not compiled, not run on hardware
//
#include <hip/hip_runtime.h>
#include <math.h>

typedef __attribute__((ext_vector_type(16))) __bf16 v16bf;
typedef __attribute__((ext_vector_type(8)))  float  v8f;

union FragU { v16bf v; unsigned int u[8]; uint4 q[2]; };

__device__ inline v8f zero8() {
  v8f z;
#pragma unroll
  for (int i = 0; i < 8; ++i) z[i] = 0.0f;
  return z;
}

__device__ inline v8f wmma_bf16(v16bf a, v16bf b, v8f c) {
  // v_wmma_f32_16x16x32_bf16  (neg_a, A, neg_b, B, c_mod, C, reuse_a, reuse_b)
  return __builtin_amdgcn_wmma_f32_16x16x32_bf16(false, a, false, b, (short)0, c,
                                                 false, false);
}

// ---- gfx1250 async global->LDS DMA (16B per lane, ASYNCcnt-tracked) -------
__device__ inline void async_ld16(const __bf16* lds_ptr, const __bf16* gptr) {
  const unsigned lds_off = (unsigned)(size_t)lds_ptr;        // LDS offset in addr[31:0]
  const unsigned long long ga = (unsigned long long)(size_t)gptr;
  asm volatile("global_load_async_to_lds_b128 %0, %1, off"
               :: "v"(lds_off), "v"(ga)
               : "memory");
}
#define ASYNC_WAIT0() asm volatile("s_wait_asynccnt 0x0" ::: "memory")
#define ASYNC_WAIT4() asm volatile("s_wait_asynccnt 0x4" ::: "memory")

// ---- Fragment loaders from LDS --------------------------------------------
// A matrix (M=16, K=32), LDS row-major [row][K] with row stride ldk (bf16).
// ISA layout: lanes 0-15 row M=lane hold K {0..7,16..23}; lanes 16-31 same row
// hold K {8..15,24..31}; packed as bf16 pairs per VGPR.
__device__ inline v16bf load_A(const __bf16* lds, int ldk, int r0, int k0) {
  const int lane = threadIdx.x & 31;
  const int row  = r0 + (lane & 15);
  const int koff = (lane >> 4) << 3; // 0 or 8
  const unsigned int* p = (const unsigned int*)(lds + row * ldk + k0);
  FragU f;
#pragma unroll
  for (int j = 0; j < 4; ++j) {
    f.u[j]     = p[(koff >> 1) + j];      // K = koff + 2j, +1
    f.u[4 + j] = p[8 + (koff >> 1) + j];  // K = 16 + koff + 2j, +1
  }
  return f.v;
}

// B matrix (K=32, N=16), LDS stored as [N][K] row-major with row stride ldk.
// ISA layout: lanes 0-15 col N=lane hold K {0..7,16..23}; lanes 16-31 hold K {8..15,24..31}.
__device__ inline v16bf load_B(const __bf16* lds, int ldk, int n0, int k0) {
  const int lane  = threadIdx.x & 31;
  const int col   = n0 + (lane & 15);
  const int khalf = (lane >> 4) << 3; // 0 or 8
  const uint4* p = (const uint4*)(lds + col * ldk + k0 + khalf);
  FragU f;
  f.q[0] = p[0];
  f.q[1] = p[2];
  return f.v;
}

__device__ inline float redmax16(float v) {
#pragma unroll
  for (int m = 1; m <= 8; m <<= 1) v = fmaxf(v, __shfl_xor(v, m, 32));
  return v;
}
__device__ inline float redsum16(float v) {
#pragma unroll
  for (int m = 1; m <= 8; m <<= 1) v += __shfl_xor(v, m, 32);
  return v;
}

// ---- K0: gate combine ------------------------------------------------------
__global__ void gate_kernel(const int* __restrict__ expert_idx,
                            const int* __restrict__ sel,
                            const float* __restrict__ gw,
                            float* __restrict__ w) {
  const int b = threadIdx.x;
  if (b < 16) {
    const int e = *expert_idx;
    float acc = 0.0f;
#pragma unroll
    for (int k = 0; k < 2; ++k)
      if (sel[b * 2 + k] == e) acc += gw[b * 2 + k];
    w[b] = acc;
  }
}

// ---- K1: f32 -> bf16 convert ----------------------------------------------
__global__ void cvt_kernel(const float* __restrict__ src, __bf16* __restrict__ dst, int n) {
  int i = blockIdx.x * blockDim.x + threadIdx.x;
  const int stride = gridDim.x * blockDim.x;
  for (; i < n; i += stride) dst[i] = (__bf16)src[i];
}

__global__ void zero_kernel(float* __restrict__ dst, int n) {
  int i = blockIdx.x * blockDim.x + threadIdx.x;
  const int stride = gridDim.x * blockDim.x;
  for (; i < n; i += stride) dst[i] = 0.0f;
}

// ---- K2/K4: WMMA GEMM, async-DMA double-buffered ---------------------------
// C[8192 x N] = X[8192 x 1024] * W^T (W is [N][1024] row-major) + bias
// Block tile 128x128, 8 waves (4x2), each wave 32x64 (2x4 WMMA tiles).
// mode 0: N=3072, X selected per n-third (query/key/value), bf16 scatter into
//         per-(b,h) [L,d] tiles.
// mode 1: N=1024, f32 output scaled by w[b].
__global__ __launch_bounds__(256) void gemm_kernel(
    const __bf16* __restrict__ X0, const __bf16* __restrict__ X1,
    const __bf16* __restrict__ X2, const __bf16* __restrict__ W,
    const float* __restrict__ bias,
    __bf16* __restrict__ q_out, __bf16* __restrict__ k_out,
    __bf16* __restrict__ v_out,
    float* __restrict__ f_out, const float* __restrict__ wgt, int mode) {
  __shared__ alignas(16) __bf16 Alds[2][128 * 32];
  __shared__ alignas(16) __bf16 Blds[2][128 * 32];

  const int tid  = threadIdx.x;
  const int lane = tid & 31, wave = tid >> 5;
  const int mt = blockIdx.x & 63, nt = blockIdx.x >> 6;
  const int m0 = mt << 7, n0 = nt << 7;

  const int third = n0 >> 10;
  const __bf16* X = (third == 0) ? X0 : (third == 1) ? X1 : X2;

  v8f acc[8];
#pragma unroll
  for (int i = 0; i < 8; ++i) acc[i] = zero8();

  const int wm = (wave >> 1) << 5; // 0,32,64,96
  const int wn = (wave & 1) << 6;  // 0,64

  // 128x32 tile = 512 x 16B segments; 256 threads x 2 segments each.
  const int sa0 = tid, sa1 = tid + 256;

  auto issueTile = [&](int buf, int kk) {
#pragma unroll
    for (int ss = 0; ss < 2; ++ss) {
      const int s = ss ? sa1 : sa0;
      const int row = s >> 2, off = (s & 3) << 3;
      async_ld16(&Alds[buf][row * 32 + off],
                 X + (size_t)(m0 + row) * 1024 + kk + off);
      async_ld16(&Blds[buf][row * 32 + off],
                 W + (size_t)(n0 + row) * 1024 + kk + off);
    }
  };

  issueTile(0, 0);
  for (int step = 0; step < 32; ++step) {
    const int kk = step << 5;
    const int cur = step & 1;
    if (step < 31) {
      issueTile(1 - cur, kk + 32);
      ASYNC_WAIT4(); // in-order completion: current buffer's 4 ops are done
    } else {
      ASYNC_WAIT0();
    }
    __syncthreads();

    v16bf a0 = load_A(Alds[cur], 32, wm, 0);
    v16bf a1 = load_A(Alds[cur], 32, wm + 16, 0);
#pragma unroll
    for (int j = 0; j < 4; ++j) {
      v16bf bj = load_B(Blds[cur], 32, wn + (j << 4), 0);
      acc[j]     = wmma_bf16(a0, bj, acc[j]);
      acc[4 + j] = wmma_bf16(a1, bj, acc[4 + j]);
    }
    __syncthreads(); // all waves done with 'cur' before it is refilled
  }

#pragma unroll
  for (int i = 0; i < 2; ++i)
#pragma unroll
    for (int j = 0; j < 4; ++j) {
      v8f c = acc[i * 4 + j];
      const int gn = n0 + wn + (j << 4) + (lane & 15);
      const float bn = bias[gn];
#pragma unroll
      for (int r = 0; r < 8; ++r) {
        const int gm = m0 + wm + (i << 4) + r + ((lane >> 4) << 3);
        const float val = c[r] + bn;
        if (mode == 0) {
          const int nn = gn & 1023, h = nn >> 6, dd = nn & 63;
          const int l = gm >> 4, bb = gm & 15;
          __bf16* dst = (third == 0) ? q_out : (third == 1) ? k_out : v_out;
          dst[(size_t)(((bb << 4) + h) * 512 + l) * 64 + dd] = (__bf16)val;
        } else {
          f_out[(size_t)gm * 1024 + gn] = val * wgt[gm & 15];
        }
      }
    }
}

// ---- K3: attention (flash-style 2-pass, WMMA) ------------------------------
// block = (b,h, 64-row l-tile); 4 waves, each owns 16 rows.
__global__ __launch_bounds__(128) void attn_kernel(
    const __bf16* __restrict__ qP, const __bf16* __restrict__ kP,
    const __bf16* __restrict__ vP, const float* __restrict__ wgt,
    float* __restrict__ out2, __bf16* __restrict__ ctxB) {
  __shared__ alignas(16) __bf16 qlds[64 * 64];
  __shared__ alignas(16) __bf16 klds[64 * 64];
  __shared__ alignas(16) __bf16 vlds[64 * 64]; // transposed: [d][s]
  __shared__ alignas(16) __bf16 plds[64 * 64];

  const int tid = threadIdx.x, lane = tid & 31, wave = tid >> 5;
  const int bh = blockIdx.x >> 3, lt = blockIdx.x & 7;
  const int b = bh >> 4, h = bh & 15;
  const int l0 = lt << 6;
  const size_t base = (size_t)bh * 512 * 64;

  // async-load q tile [64 x 64] : 512 x 16B segments, 128 threads x 4 each
#pragma unroll
  for (int e = 0; e < 4; ++e) {
    const int s = tid + (e << 7);
    const int row = s >> 3, off = (s & 7) << 3;
    async_ld16(&qlds[row * 64 + off], qP + base + (size_t)(l0 + row) * 64 + off);
  }
  ASYNC_WAIT0();
  __syncthreads();

  float m8[8], z8[8];
#pragma unroll
  for (int r = 0; r < 8; ++r) { m8[r] = -3.0e38f; z8[r] = 0.0f; }

  // ---- pass 1: softmax statistics (online max/sum)
  for (int sc = 0; sc < 8; ++sc) {
    const int s0 = sc << 6;
#pragma unroll
    for (int e = 0; e < 4; ++e) {
      const int s = tid + (e << 7);
      const int row = s >> 3, off = (s & 7) << 3;
      async_ld16(&klds[row * 64 + off], kP + base + (size_t)(s0 + row) * 64 + off);
    }
    ASYNC_WAIT0();
    __syncthreads();
#pragma unroll
    for (int nt = 0; nt < 4; ++nt) {
      v8f sv = zero8();
#pragma unroll
      for (int kb = 0; kb < 64; kb += 32)
        sv = wmma_bf16(load_A(qlds, 64, wave << 4, kb), load_B(klds, 64, nt << 4, kb), sv);
#pragma unroll
      for (int r = 0; r < 8; ++r) {
        const float s = sv[r] * 0.125f;
        const float mn = fmaxf(m8[r], redmax16(s));
        const float ze = redsum16(__expf(s - mn));
        z8[r] = z8[r] * __expf(m8[r] - mn) + ze;
        m8[r] = mn;
      }
    }
    __syncthreads();
  }

  v8f ctx[4];
#pragma unroll
  for (int d = 0; d < 4; ++d) ctx[d] = zero8();
  const float winv = wgt[b] * 0.0625f; // w[b] / NUM_HEADS

  // ---- pass 2: P = softmax, avg-attn scatter, ctx = P x V
  for (int sc = 0; sc < 8; ++sc) {
    const int s0 = sc << 6;
#pragma unroll
    for (int e = 0; e < 4; ++e) {
      const int s = tid + (e << 7);
      const int row = s >> 3, off = (s & 7) << 3;
      async_ld16(&klds[row * 64 + off], kP + base + (size_t)(s0 + row) * 64 + off);
    }
#pragma unroll
    for (int e = 0; e < 32; ++e) { // v chunk, transposed into LDS
      const int idx = tid * 32 + e;
      const int s = idx >> 6, d = idx & 63;
      vlds[d * 64 + s] = vP[base + (size_t)(s0 + s) * 64 + d];
    }
    ASYNC_WAIT0();
    __syncthreads();
#pragma unroll
    for (int nt = 0; nt < 4; ++nt) {
      v8f sv = zero8();
#pragma unroll
      for (int kb = 0; kb < 64; kb += 32)
        sv = wmma_bf16(load_A(qlds, 64, wave << 4, kb), load_B(klds, 64, nt << 4, kb), sv);
      const int scol = s0 + (nt << 4) + (lane & 15);
#pragma unroll
      for (int r = 0; r < 8; ++r) {
        const float p = __expf(sv[r] * 0.125f - m8[r]) / z8[r];
        const int lrow = l0 + (wave << 4) + r + ((lane >> 4) << 3);
        unsafeAtomicAdd(out2 + (size_t)b * 262144 + (size_t)lrow * 512 + scol, p * winv);
        plds[((wave << 4) + r + ((lane >> 4) << 3)) * 64 + (nt << 4) + (lane & 15)] = (__bf16)p;
      }
    }
    // ctx += P[16x64] x V[64x64]   (per-wave rows; LDS in-order per wave)
#pragma unroll
    for (int dt = 0; dt < 4; ++dt)
#pragma unroll
      for (int kb = 0; kb < 64; kb += 32)
        ctx[dt] = wmma_bf16(load_A(plds, 64, wave << 4, kb), load_B(vlds, 64, dt << 4, kb), ctx[dt]);
    __syncthreads();
  }

#pragma unroll
  for (int dt = 0; dt < 4; ++dt)
#pragma unroll
    for (int r = 0; r < 8; ++r) {
      const int lrow = l0 + (wave << 4) + r + ((lane >> 4) << 3);
      ctxB[(size_t)(lrow * 16 + b) * 1024 + h * 64 + (dt << 4) + (lane & 15)] =
          (__bf16)ctx[dt][r];
    }
}

// ---------------------------------------------------------------------------
extern "C" void kernel_launch(void* const* d_in, const int* in_sizes, int n_in,
                              void* d_out, int out_size, void* d_ws, size_t ws_size,
                              hipStream_t stream) {
  const int*   expert_idx = (const int*)d_in[0];
  const int*   gate_sel   = (const int*)d_in[1];
  const float* gate_w     = (const float*)d_in[2];
  const float* query      = (const float*)d_in[3];
  const float* key        = (const float*)d_in[4];
  const float* value      = (const float*)d_in[5];
  const float* Win        = (const float*)d_in[6];
  const float* bin        = (const float*)d_in[7];
  const float* Wout       = (const float*)d_in[8];
  const float* bout       = (const float*)d_in[9];

  float* out1 = (float*)d_out;            // [512,16,1024]
  float* out2 = out1 + 8388608;           // [16,512,512]

  char* ws = (char*)d_ws;
  size_t off = 0;
  auto take = [&](size_t bytes) -> void* {
    void* p = ws + off;
    off += (bytes + 255) & ~(size_t)255;
    return p;
  };
  float*  wgt   = (float*)take(16 * sizeof(float));
  __bf16* xq    = (__bf16*)take((size_t)8192 * 1024 * 2);
  __bf16* xk    = (__bf16*)take((size_t)8192 * 1024 * 2);
  __bf16* xv    = (__bf16*)take((size_t)8192 * 1024 * 2);
  __bf16* WinB  = (__bf16*)take((size_t)3072 * 1024 * 2);
  __bf16* WoutB = (__bf16*)take((size_t)1024 * 1024 * 2);
  __bf16* qP    = (__bf16*)take((size_t)256 * 512 * 64 * 2);
  __bf16* kP    = (__bf16*)take((size_t)256 * 512 * 64 * 2);
  __bf16* vP    = (__bf16*)take((size_t)256 * 512 * 64 * 2);
  __bf16* ctxB  = (__bf16*)take((size_t)8192 * 1024 * 2);

  gate_kernel<<<1, 32, 0, stream>>>(expert_idx, gate_sel, gate_w, wgt);
  cvt_kernel<<<2048, 256, 0, stream>>>(query, xq, 8388608);
  cvt_kernel<<<2048, 256, 0, stream>>>(key, xk, 8388608);
  cvt_kernel<<<2048, 256, 0, stream>>>(value, xv, 8388608);
  cvt_kernel<<<2048, 256, 0, stream>>>(Win, WinB, 3145728);
  cvt_kernel<<<1024, 256, 0, stream>>>(Wout, WoutB, 1048576);
  zero_kernel<<<2048, 256, 0, stream>>>(out2, 4194304);

  // QKV projection: M=8192, N=3072 -> 64 x 24 tiles of 128x128
  gemm_kernel<<<64 * 24, 256, 0, stream>>>(xq, xk, xv, WinB, bin,
                                           qP, kP, vP, nullptr, nullptr, 0);
  // Attention: (b,h) x 8 l-tiles
  attn_kernel<<<2048, 128, 0, stream>>>(qP, kP, vP, wgt, out2, ctxB);
  // Out projection: M=8192, N=1024 -> 64 x 8 tiles of 128x128
  gemm_kernel<<<64 * 8, 256, 0, stream>>>(ctxB, ctxB, ctxB, WoutB, bout,
                                          nullptr, nullptr, nullptr, out1, wgt, 1);
}
